// InterAttention_24713241821781
// MI455X (gfx1250) — matrix operand order repaired and checked by reading, not compiled, not run on hardware
//
#include <hip/hip_runtime.h>
#include <hip/hip_bf16.h>

typedef __attribute__((ext_vector_type(16))) _Float16 v16h;
typedef __attribute__((ext_vector_type(8)))  _Float16 v8h;
typedef __attribute__((ext_vector_type(8)))  float    v8f;
typedef __attribute__((ext_vector_type(4)))  float    v4f;

#define HEADS 8
#define DHEAD 64
#define BATCH 8
#define SEQ   1024
#define DIM   512
#define INNER 512           // HEADS*DHEAD
#define SCALE 0.125f        // 64^-0.5
#define KK    256           // top-k = SEQ/4

// ---------------------------------------------------------------------------
// Fragment loaders (WMMA 16x16x32 f16; contiguous k order, same in A and B)
// A (16xK=32): lane m = l&15; halves 0-15 = K[kb..kb+15],
//              kb = (l>>4)*16
// ---------------------------------------------------------------------------
__device__ inline v16h load_a_f16(const _Float16* p, int stride) {
  int l = threadIdx.x & 31;
  const _Float16* q = p + (size_t)(l & 15) * stride + ((l >> 4) << 4);
  v8h lo = *(const v8h*)q;
  v8h hi = *(const v8h*)(q + 8);
  v16h a;
#pragma unroll
  for (int i = 0; i < 8; ++i) { a[i] = lo[i]; a[i + 8] = hi[i]; }
  return a;
}

// B (K=32 x 16), sourced from B^T stored row-major [n][k]:
// lane n = l&15; 16 contiguous K halves starting at (l>>4)*16
__device__ inline v16h load_bt16(const _Float16* p, int stride) {
  int l = threadIdx.x & 31;
  const _Float16* q = p + (size_t)(l & 15) * stride + ((l >> 4) << 4);
  v8h lo = *(const v8h*)q;
  v8h hi = *(const v8h*)(q + 8);
  v16h b;
#pragma unroll
  for (int i = 0; i < 8; ++i) { b[i] = lo[i]; b[i + 8] = hi[i]; }
  return b;
}

__device__ inline unsigned f2ord(float f) {
  unsigned u = __float_as_uint(f);
  return (u & 0x80000000u) ? ~u : (u | 0x80000000u);
}
__device__ inline float ord2f(unsigned u) {
  return __uint_as_float((u & 0x80000000u) ? (u ^ 0x80000000u) : ~u);
}

// ---------------------------------------------------------------------------
// Kernel: weight transpose + f32->f16   W[K][N] -> Wt[N][K]
// ---------------------------------------------------------------------------
__global__ void wt_kernel(const float* __restrict__ W, _Float16* __restrict__ Wt,
                          int K, int N) {
  int idx = blockIdx.x * 256 + threadIdx.x;
  if (idx < K * N) {
    int n = idx / K, k = idx % K;
    Wt[idx] = (_Float16)W[(size_t)k * N + n];
  }
}

// ---------------------------------------------------------------------------
// Kernel: activation f32 -> f16 (8 elements per thread, vectorized)
// ---------------------------------------------------------------------------
__global__ __launch_bounds__(256) void xcvt_kernel(const float* __restrict__ X,
                                                   _Float16* __restrict__ Xh) {
  size_t base = ((size_t)blockIdx.x * 256 + threadIdx.x) * 8;
  v4f lo = *(const v4f*)(X + base);
  v4f hi = *(const v4f*)(X + base + 4);
  v8h o;
#pragma unroll
  for (int i = 0; i < 4; ++i) { o[i] = (_Float16)lo[i]; o[i + 4] = (_Float16)hi[i]; }
  *(v8h*)(Xh + base) = o;
}

// ---------------------------------------------------------------------------
// Generic 16x64-per-wave GEMM body: C[16][64] += A[16][512] * Bt[64][512]^T
// ---------------------------------------------------------------------------
__device__ inline void gemm_16x64(const _Float16* ab, const _Float16* bb,
                                  v8f acc[4]) {
  int lane = threadIdx.x & 31;
  for (int kt = 0; kt < DIM / 32; ++kt) {
    if (kt + 1 < DIM / 32) {
      __builtin_prefetch(ab + (size_t)(lane & 15) * DIM + (kt + 1) * 32, 0, 0);
      __builtin_prefetch(bb + (size_t)(lane & 15) * DIM + (kt + 1) * 32, 0, 0);
    }
    v16h a = load_a_f16(ab + kt * 32, DIM);
#pragma unroll
    for (int s = 0; s < 4; ++s) {
      v16h b = load_bt16(bb + (size_t)s * 16 * DIM + kt * 32, DIM);
      acc[s] = __builtin_amdgcn_wmma_f32_16x16x32_f16(false, a, false, b,
                                                      (short)0, acc[s], false,
                                                      false);
    }
  }
}

// ---------------------------------------------------------------------------
// Kernel: Q projection   Q = Xh @ wq  (16x64 tile per wave, 8 waves/block)
// ---------------------------------------------------------------------------
__global__ __launch_bounds__(256) void proj_q_kernel(
    const _Float16* __restrict__ Xh, const _Float16* __restrict__ Wt,
    _Float16* __restrict__ Qd) {
  int wave = threadIdx.x >> 5, lane = threadIdx.x & 31;
  int tile = blockIdx.x * 8 + wave;
  int tn = tile & 7;           // 512/64 N slabs
  int tm = tile >> 3;          // 8192/16 M tiles
  v8f acc[4] = {};
  gemm_16x64(Xh + (size_t)tm * 16 * DIM, Wt + (size_t)tn * 64 * DIM, acc);
  int n = lane & 15, mb = (lane >> 4) * 8;
#pragma unroll
  for (int s = 0; s < 4; ++s) {
    _Float16* od = Qd + (size_t)(tm * 16 + mb) * INNER + tn * 64 + s * 16 + n;
#pragma unroll
    for (int r = 0; r < 8; ++r) od[(size_t)r * INNER] = (_Float16)acc[s][r];
  }
}

// ---------------------------------------------------------------------------
// Kernel: KV projection  [K|V] = Xh @ wkv. K stored [row][512];
// V stored transposed Vt[(b*512 + h*64 + d)][j] for contiguous PV B-fragments.
// ---------------------------------------------------------------------------
__global__ __launch_bounds__(256) void proj_kv_kernel(
    const _Float16* __restrict__ Xh, const _Float16* __restrict__ Wt,
    _Float16* __restrict__ Kd, _Float16* __restrict__ Vtd) {
  int wave = threadIdx.x >> 5, lane = threadIdx.x & 31;
  int tile = blockIdx.x * 8 + wave;
  int tn = tile & 15;          // 1024/64 N slabs
  int tm = tile >> 4;          // 8192/16 M tiles
  v8f acc[4] = {};
  gemm_16x64(Xh + (size_t)tm * 16 * DIM, Wt + (size_t)tn * 64 * DIM, acc);
  int n = lane & 15, mb = (lane >> 4) * 8;
#pragma unroll
  for (int s = 0; s < 4; ++s) {
    int col = tn * 64 + s * 16 + n;     // (col<512) uniform per sub-tile
    if (col < INNER) {
#pragma unroll
      for (int r = 0; r < 8; ++r) {
        int rowg = tm * 16 + mb + r;
        Kd[(size_t)rowg * INNER + col] = (_Float16)acc[s][r];
      }
    } else {
      int c2 = col - INNER;             // h*64 + d
#pragma unroll
      for (int r = 0; r < 8; ++r) {
        int rowg = tm * 16 + mb + r;
        int b_ = rowg >> 10, j = rowg & 1023;
        Vtd[((size_t)(b_ * INNER + c2)) * SEQ + j] = (_Float16)acc[s][r];
      }
    }
  }
}

// ---------------------------------------------------------------------------
// Kernel: attention per (b, h, 16-query tile). 128 threads = 4 waves.
// dyn LDS: S[16][1024] f32, P[16][1024] f16, hist[16][256] i32, rowsum[16]
// ---------------------------------------------------------------------------
__global__ __launch_bounds__(128) void attn_kernel(
    const _Float16* __restrict__ Q,   // [b*n][512]
    const _Float16* __restrict__ Km,  // [b*n][512]
    const _Float16* __restrict__ Vt,  // [(b*512 + h*64 + d)][n]
    _Float16* __restrict__ O) {       // [b*n][512] merged heads
  extern __shared__ char smem[];
  float*    S    = (float*)smem;                                 // 64 KB
  _Float16* P    = (_Float16*)(smem + 16 * SEQ * 4);             // 32 KB
  int*      hist = (int*)(smem + 16 * SEQ * 4 + 16 * SEQ * 2);   // 16 KB
  float* rowsum  = (float*)(smem + 16 * SEQ * 4 + 16 * SEQ * 2 + 16 * 256 * 4);

  int tid = threadIdx.x, lane = tid & 31, wave = tid >> 5;
  int bh = blockIdx.x >> 6, qt = blockIdx.x & 63;
  int b = bh >> 3, h = bh & 7;

  // --- S = scale * Q Kt for this wave's 256-column slab ---
  const _Float16* qbase = Q + (size_t)(b * SEQ + qt * 16) * INNER + h * DHEAD;
  v16h a0 = load_a_f16(qbase, INNER);
  v16h a1 = load_a_f16(qbase + 32, INNER);
  for (int jt = 0; jt < 16; ++jt) {
    int jb = wave * 256 + jt * 16;
    const _Float16* kb = Km + (size_t)(b * SEQ + jb) * INNER + h * DHEAD;
    if (jt < 15)  // prefetch next K tile rows (128B per row)
      __builtin_prefetch(kb + (size_t)(16 + (lane & 15)) * INNER, 0, 0);
    v16h b0 = load_bt16(kb, INNER);
    v16h b1 = load_bt16(kb + 32, INNER);
    v8f c = {};
    c = __builtin_amdgcn_wmma_f32_16x16x32_f16(false, a0, false, b0, (short)0, c,
                                               false, false);
    c = __builtin_amdgcn_wmma_f32_16x16x32_f16(false, a1, false, b1, (short)0, c,
                                               false, false);
    int n = lane & 15, mb = (lane >> 4) * 8;
#pragma unroll
    for (int r = 0; r < 8; ++r) S[(mb + r) * SEQ + jb + n] = c[r] * SCALE;
  }
  __syncthreads();

  // --- per-row MSD radix select of 256th-largest (ordered-uint space) ---
  int row = tid >> 3, part = tid & 7;   // 8 lanes per row, same wave
  const float* srow = S + row * SEQ + part * 128;
  int* hrow = hist + row * 256;

  unsigned prefix = 0, umax = 0;
  int need = KK;
  for (int d = 3; d >= 0; --d) {
    int sh = d * 8;
    unsigned hmask = (d == 3) ? 0u : (0xFFFFFFFFu << (sh + 8));
#pragma unroll
    for (int i = 0; i < 32; ++i) hrow[part * 32 + i] = 0;
    __syncthreads();
    for (int j = 0; j < 128; ++j) {
      unsigned u = f2ord(srow[j]);
      if (d == 3) umax = (u > umax) ? u : umax;
      if ((u & hmask) == prefix) atomicAdd(&hrow[(u >> sh) & 255], 1);
    }
    __syncthreads();
    // all 8 lanes replicate the downward cumulative scan (same result)
    int cum = 0, chosen = 0;
    for (int bin = 255; bin >= 0; --bin) {
      int hcnt = hrow[bin];
      if (cum + hcnt >= need) { chosen = bin; break; }
      cum += hcnt;
    }
    prefix |= (unsigned)chosen << sh;
    need -= cum;
    __syncthreads();
  }
  unsigned T = prefix;  // value of 256th largest: keep u >= T
  {
    unsigned o;
    o = (unsigned)__shfl_xor((int)umax, 1); umax = umax > o ? umax : o;
    o = (unsigned)__shfl_xor((int)umax, 2); umax = umax > o ? umax : o;
    o = (unsigned)__shfl_xor((int)umax, 4); umax = umax > o ? umax : o;
  }
  float mx = ord2f(umax);

  // --- exp + sum, write unnormalized P (norm folded into O epilogue) ---
  float sum = 0.f;
  _Float16* prow = P + row * SEQ + part * 128;
  for (int j = 0; j < 128; ++j) {
    float s = srow[j];
    float p = (f2ord(s) >= T) ? __expf(s - mx) : 0.0f;
    prow[j] = (_Float16)p;
    sum += p;
  }
  sum += __shfl_xor(sum, 1);
  sum += __shfl_xor(sum, 2);
  sum += __shfl_xor(sum, 4);
  if (part == 0) rowsum[row] = sum;
  __syncthreads();

  // --- O(16 x 16d per wave) = P @ V, normalize rows in epilogue ---
  v8f acc = {};
  const _Float16* vtb = Vt + (size_t)(b * INNER + h * DHEAD + wave * 16) * SEQ;
  for (int kt = 0; kt < SEQ / 32; ++kt) {
    if (kt + 1 < SEQ / 32)
      __builtin_prefetch(vtb + (size_t)(lane & 15) * SEQ + (kt + 1) * 32, 0, 0);
    v16h pa = load_a_f16(P + kt * 32, SEQ);
    v16h vb = load_bt16(vtb + kt * 32, SEQ);
    acc = __builtin_amdgcn_wmma_f32_16x16x32_f16(false, pa, false, vb, (short)0,
                                                 acc, false, false);
  }
  int n = lane & 15, mb = (lane >> 4) * 8;
  _Float16* ob =
      O + (size_t)(b * SEQ + qt * 16) * INNER + h * DHEAD + wave * 16 + n;
#pragma unroll
  for (int r = 0; r < 8; ++r)
    ob[(size_t)(mb + r) * INNER] = (_Float16)(acc[r] / rowsum[mb + r]);
}

// ---------------------------------------------------------------------------
// Kernel: out projection  out = O @ wout + bout  (16x64 per wave, f32 out)
// ---------------------------------------------------------------------------
__global__ __launch_bounds__(256) void outproj_kernel(
    const _Float16* __restrict__ O, const _Float16* __restrict__ Wt,
    const float* __restrict__ bias, float* __restrict__ out) {
  int wave = threadIdx.x >> 5, lane = threadIdx.x & 31;
  int tile = blockIdx.x * 8 + wave;
  int tn = tile & 7, tm = tile >> 3;
  v8f acc[4] = {};
  gemm_16x64(O + (size_t)tm * 16 * INNER, Wt + (size_t)tn * 64 * INNER, acc);
  int n = lane & 15, mb = (lane >> 4) * 8;
#pragma unroll
  for (int s = 0; s < 4; ++s) {
    int col = tn * 64 + s * 16 + n;
    float bv = bias[col];
#pragma unroll
    for (int r = 0; r < 8; ++r)
      out[(size_t)(tm * 16 + mb + r) * DIM + col] = acc[s][r] + bv;
  }
}

// ---------------------------------------------------------------------------
extern "C" void kernel_launch(void* const* d_in, const int* in_sizes, int n_in,
                              void* d_out, int out_size, void* d_ws,
                              size_t ws_size, hipStream_t stream) {
  const float* x    = (const float*)d_in[0];
  const float* y    = (const float*)d_in[1];
  const float* wq   = (const float*)d_in[2];
  const float* wkv  = (const float*)d_in[3];
  const float* wout = (const float*)d_in[4];
  const float* bout = (const float*)d_in[5];
  float* out = (float*)d_out;

  char* ws = (char*)d_ws;
  size_t off = 0;
  auto alloc = [&](size_t bytes) {
    void* p = ws + off;
    off += (bytes + 255) & ~(size_t)255;
    return p;
  };
  _Float16* wq_t   = (_Float16*)alloc((size_t)INNER * DIM * 2);
  _Float16* wkv_t  = (_Float16*)alloc((size_t)2 * INNER * DIM * 2);
  _Float16* wout_t = (_Float16*)alloc((size_t)DIM * INNER * 2);
  const size_t tsz = (size_t)BATCH * SEQ * INNER * 2;  // 8 MB each
  _Float16* Xh  = (_Float16*)alloc(tsz);
  _Float16* Yh  = (_Float16*)alloc(tsz);
  _Float16* Qx  = (_Float16*)alloc(tsz);
  _Float16* Qy  = (_Float16*)alloc(tsz);
  _Float16* Kx  = (_Float16*)alloc(tsz);
  _Float16* Ky  = (_Float16*)alloc(tsz);
  _Float16* Vtx = (_Float16*)alloc(tsz);
  _Float16* Vty = (_Float16*)alloc(tsz);
  _Float16* Ox  = (_Float16*)alloc(tsz);
  _Float16* Oy  = (_Float16*)alloc(tsz);

  // 1) weight transpose/convert + activation convert
  wt_kernel<<<(DIM * INNER + 255) / 256, 256, 0, stream>>>(wq, wq_t, DIM, INNER);
  wt_kernel<<<(DIM * 2 * INNER + 255) / 256, 256, 0, stream>>>(wkv, wkv_t, DIM,
                                                               2 * INNER);
  wt_kernel<<<(INNER * DIM + 255) / 256, 256, 0, stream>>>(wout, wout_t, INNER,
                                                           DIM);
  xcvt_kernel<<<2048, 256, 0, stream>>>(x, Xh);
  xcvt_kernel<<<2048, 256, 0, stream>>>(y, Yh);

  // 2) projections (M=8192 rows, 16x64 per wave, 8 waves/block)
  proj_q_kernel<<<512, 256, 0, stream>>>(Xh, wq_t, Qx);
  proj_q_kernel<<<512, 256, 0, stream>>>(Yh, wq_t, Qy);
  proj_kv_kernel<<<1024, 256, 0, stream>>>(Xh, wkv_t, Kx, Vtx);
  proj_kv_kernel<<<1024, 256, 0, stream>>>(Yh, wkv_t, Ky, Vty);

  // 3) attention: out_x = attn(Qx vs y), out_y = attn(Qy vs x)
  size_t attn_smem = (size_t)16 * SEQ * 4 + (size_t)16 * SEQ * 2 +
                     (size_t)16 * 256 * 4 + 256;
  hipFuncSetAttribute((const void*)attn_kernel,
                      hipFuncAttributeMaxDynamicSharedMemorySize,
                      (int)attn_smem);
  attn_kernel<<<BATCH * HEADS * (SEQ / 16), 128, attn_smem, stream>>>(Qx, Ky,
                                                                      Vty, Ox);
  attn_kernel<<<BATCH * HEADS * (SEQ / 16), 128, attn_smem, stream>>>(Qy, Kx,
                                                                      Vtx, Oy);

  // 4) output projection (out_x first, then out_y)
  outproj_kernel<<<512, 256, 0, stream>>>(Ox, wout_t, bout, out);
  outproj_kernel<<<512, 256, 0, stream>>>(Oy, wout_t, bout,
                                          out + (size_t)BATCH * SEQ * DIM);
  (void)in_sizes; (void)n_in; (void)out_size; (void)ws_size;
}
